// SpikingSelfAttention_87969520156927
// MI455X (gfx1250) — matrix operand order repaired and checked by reading, not compiled, not run on hardware
//
#include <hip/hip_runtime.h>

// ---------------------------------------------------------------------------
// Spiking self-attention (T=4,B=32,C=384,N=196,12 heads x 32) for gfx1250.
// All GEMMs routed through v_wmma_f32_16x16x32_f16 (spikes are exact in f16).
// Scratch tensors are zero-padded (N 196->208 rows / 196->224 cols) so all
// WMMA fragment loads are unguarded, aligned global_load_b128 (no EXEC masks
// or zero-fill movs inside the K loops).
// ---------------------------------------------------------------------------

typedef __attribute__((ext_vector_type(16))) _Float16 v16h;
typedef __attribute__((ext_vector_type(8)))  _Float16 v8h;
typedef __attribute__((ext_vector_type(8)))  float    v8f;

namespace {
constexpr int    T_  = 4;
constexpr int    B_  = 32;
constexpr int    C_  = 384;
constexpr int    NH  = 12;   // heads
constexpr int    HD  = 32;   // head dim
constexpr int    N_  = 196;  // 14*14
constexpr int    NPn = 208;  // n padded to 13*16 (for [T,B,NPn,C] layouts)
constexpr int    NPc = 224;  // n padded to 7*32  (for [T,B,C,NPc] layouts)
constexpr size_t BCN  = (size_t)B_ * C_ * N_;   // 2,408,448
constexpr size_t TBCN = (size_t)T_ * BCN;       // 9,633,792
constexpr int    OT  = C_ / 16;                 // 24 output-channel tiles
constexpr int    NT  = NPn / 16;                // 13 spatial tiles
constexpr float  EPSV = 1e-5f;
}

__device__ __forceinline__ v8f wmma_f16(v16h a, v16h b, v8f c) {
  // D = A(16x32 f16) * B(32x16 f16) + C(16x16 f32)
  return __builtin_amdgcn_wmma_f32_16x16x32_f16(false, a, false, b, (short)0, c,
                                                false, false);
}

// A fragment, row-major f16, 16B-aligned rows of `stride` halfs.
// ISA 7.12.2 (16-bit A 16x32): lane<16 -> K {0..7,16..23},
// lane>=16 -> K {8..15,24..31}, M = lane&15.
__device__ __forceinline__ v16h load_a_frag(const _Float16* __restrict__ p) {
  const v8h* pv = (const v8h*)p;
  v8h lo = pv[0];
  v8h up = pv[2];  // +16 halfs
  v16h a;
#pragma unroll
  for (int i = 0; i < 8; ++i) { a[i] = lo[i]; a[i + 8] = up[i]; }
  return a;
}

// B fragment: lane holds column (lane&15), K same as A fragment
// (lane<16 -> K {0..7,16..23}, lane>=16 -> K {8..15,24..31}). p must be 16B aligned.
__device__ __forceinline__ v16h load_b_frag(const _Float16* __restrict__ p) {
  const v8h* pv = (const v8h*)p;
  v8h b0 = pv[0];
  v8h b1 = pv[2];
  v16h b;
#pragma unroll
  for (int i = 0; i < 8; ++i) { b[i] = b0[i]; b[i + 8] = b1[i]; }
  return b;
}

// ------------------------- pad zero-fill -----------------------------------
// [T,B,NPn,C]: zero rows n = 196..207.
__global__ void zero_pad_nc_kernel(_Float16* __restrict__ buf)
{
  int idx = blockIdx.x * blockDim.x + threadIdx.x;   // < T*B*12*C = 589824
  int c  = idx % C_;
  int r  = (idx / C_) % (NPn - N_);
  int tb = idx / (C_ * (NPn - N_));
  buf[((size_t)tb * NPn + N_ + r) * C_ + c] = (_Float16)0.f;
}

// [T,B,C,NPc]: zero cols n = 196..223.
__global__ void zero_pad_cn_kernel(_Float16* __restrict__ buf)
{
  int idx = blockIdx.x * blockDim.x + threadIdx.x;   // < T*B*C*28 = 1376256
  int p  = idx % (NPc - N_);
  int ch = (idx / (NPc - N_)) % C_;
  int tb = idx / ((NPc - N_) * C_);
  buf[((size_t)tb * C_ + ch) * NPc + N_ + p] = (_Float16)0.f;
}

// ------------------------------ prep ---------------------------------------
// f32->f16 weight conversion + BN constant folding.
__global__ void prep_kernel(
    const float* __restrict__ qw, const float* __restrict__ kw,
    const float* __restrict__ vw, const float* __restrict__ pw,
    const float* __restrict__ q_s, const float* __restrict__ q_b,
    const float* __restrict__ q_m, const float* __restrict__ q_v,
    const float* __restrict__ k_s, const float* __restrict__ k_b,
    const float* __restrict__ k_m, const float* __restrict__ k_v,
    const float* __restrict__ v_s, const float* __restrict__ v_b,
    const float* __restrict__ v_m, const float* __restrict__ v_v,
    const float* __restrict__ proj_b, const float* __restrict__ p_s,
    const float* __restrict__ p_b, const float* __restrict__ p_m,
    const float* __restrict__ p_v,
    _Float16* __restrict__ whq, _Float16* __restrict__ whk,
    _Float16* __restrict__ whv, _Float16* __restrict__ whp,
    float* __restrict__ cst /* 8 x 384: invq,betaq,invk,betak,invv,betav,invp,betap */)
{
  int idx = blockIdx.x * blockDim.x + threadIdx.x;
  const int WSZ = C_ * C_;
  if (idx < 4 * WSZ) {
    int w = idx / WSZ, e = idx % WSZ;
    const float* src = (w == 0) ? qw : (w == 1) ? kw : (w == 2) ? vw : pw;
    _Float16*    dst = (w == 0) ? whq : (w == 1) ? whk : (w == 2) ? whv : whp;
    dst[e] = (_Float16)src[e];
  } else if (idx < 4 * WSZ + 4 * C_) {
    int j = idx - 4 * WSZ;
    int set = j / C_, c = j % C_;
    if (set == 0) {
      float iv = q_s[c] * rsqrtf(q_v[c] + EPSV);
      cst[0 * C_ + c] = iv; cst[1 * C_ + c] = q_b[c] - q_m[c] * iv;
    } else if (set == 1) {
      float iv = k_s[c] * rsqrtf(k_v[c] + EPSV);
      cst[2 * C_ + c] = iv; cst[3 * C_ + c] = k_b[c] - k_m[c] * iv;
    } else if (set == 2) {
      float iv = v_s[c] * rsqrtf(v_v[c] + EPSV);
      cst[4 * C_ + c] = iv; cst[5 * C_ + c] = v_b[c] - v_m[c] * iv;
    } else {
      float iv = p_s[c] * rsqrtf(p_v[c] + EPSV);
      cst[6 * C_ + c] = iv;
      cst[7 * C_ + c] = iv * (proj_b[c] - p_m[c]) + p_b[c];  // fold conv bias + BN
    }
  }
}

// ------------------------------ lif(x) -------------------------------------
// Elementwise T-scan; emits f32 spikes (output slice) + f16 copy in [T,B,NPn,C].
__global__ void lif_x_kernel(const float* __restrict__ x,
                             float* __restrict__ xs_f,
                             _Float16* __restrict__ xs_h)
{
  size_t idx = (size_t)blockIdx.x * blockDim.x + threadIdx.x;
  if (idx >= BCN) return;
  int n = (int)(idx % N_);
  int c = (int)((idx / N_) % C_);
  int b = (int)(idx / ((size_t)C_ * N_));
  float v = 0.f;
#pragma unroll
  for (int t = 0; t < T_; ++t) {
    float xv = x[(size_t)t * BCN + idx];
    v = 0.5f * (v + xv);                        // tau = 2
    float s = (v >= 1.0f) ? 1.0f : 0.0f;
    xs_f[(size_t)t * BCN + idx] = s;
    xs_h[(((size_t)t * B_ + b) * NPn + n) * C_ + c] = (_Float16)s;
    v *= (1.0f - s);
  }
}

// --------------------- q/k/v: 1x1 conv + BN + LIF (fused) ------------------
// One wave per 16x16 (o,n) tile; 4 accumulators (one per time step) so the
// weight fragment is reused 4x per K step. K loop: 12 x wmma_f32_16x16x32_f16.
// All fragment loads unguarded (padded scratch).
__global__ void qkv_gemm_lif_kernel(const _Float16* __restrict__ W,
                                    const _Float16* __restrict__ xs_h, // [T,B,NPn,C]
                                    const float* __restrict__ inv_c,
                                    const float* __restrict__ beta_c,
                                    float* __restrict__ sf,            // [T,B,C,N]
                                    _Float16* __restrict__ sh)         // [T,B,C,NPc]
{
  int wave = (int)((blockIdx.x * blockDim.x + threadIdx.x) >> 5);
  int lane = threadIdx.x & 31;
  int nt = wave % NT, ot = (wave / NT) % OT, b = wave / (NT * OT);
  int o0 = ot * 16, n0 = nt * 16;
  int hi = lane >> 4, lm = lane & 15;
  int n = n0 + lm;                 // may reach 207: reads padded zero rows

  v8f acc[T_];
#pragma unroll
  for (int t = 0; t < T_; ++t) acc[t] = (v8f){};

  const _Float16* wrow = W + (size_t)(o0 + lm) * C_ + hi * 8;
  const _Float16* brow[T_];
#pragma unroll
  for (int t = 0; t < T_; ++t)
    brow[t] = xs_h + (((size_t)t * B_ + b) * NPn + n) * C_ + hi * 8;

  for (int k0 = 0; k0 < C_; k0 += 32) {
    v16h a = load_a_frag(wrow + k0);
#pragma unroll
    for (int t = 0; t < T_; ++t)
      acc[t] = wmma_f16(a, load_b_frag(brow[t] + k0), acc[t]);
  }

  if (n < N_) {
#pragma unroll
    for (int r = 0; r < 8; ++r) {
      int o = o0 + r + hi * 8;        // C-layout: VGPR r -> M = r (+8 for hi lanes)
      float iv = inv_c[o], bt = beta_c[o];
      float v = 0.f;
#pragma unroll
      for (int t = 0; t < T_; ++t) {
        float y = acc[t][r] * iv + bt;
        v = 0.5f * (v + y);
        float s = (v >= 1.0f) ? 1.0f : 0.0f;
        size_t tb = (size_t)t * B_ + b;
        sf[(tb * C_ + o) * N_  + n] = s;
        sh[(tb * C_ + o) * NPc + n] = (_Float16)s;
        v *= (1.0f - s);
      }
    }
  }
}

// --------------------------- kv = k^T @ v ----------------------------------
// 32x32 per (t,b,h): 2x2 WMMA tiles, K = N padded 196 -> 224 (zeros), all
// fragments are aligned vector loads.
__global__ void kv_kernel(const _Float16* __restrict__ kc, // [T,B,C,NPc]
                          const _Float16* __restrict__ vc, // [T,B,C,NPc]
                          float* __restrict__ kvf)         // [T,B,NH,32,32]
{
  int wave = (int)((blockIdx.x * blockDim.x + threadIdx.x) >> 5);
  int lane = threadIdx.x & 31;
  int h = wave % NH, b = (wave / NH) % B_, t = wave / (NH * B_);
  int hi = lane >> 4, lm = lane & 15;
  size_t cb = ((size_t)t * B_ + b) * C_ + h * HD;

  v8f acc[2][2];
#pragma unroll
  for (int i = 0; i < 2; ++i)
#pragma unroll
    for (int j = 0; j < 2; ++j) acc[i][j] = (v8f){};

  const _Float16* arow0 = kc + (cb + lm)      * NPc + hi * 8;   // A rows d
  const _Float16* arow1 = kc + (cb + 16 + lm) * NPc + hi * 8;
  const _Float16* brow0 = vc + (cb + lm)      * NPc + hi * 8;  // B cols e
  const _Float16* brow1 = vc + (cb + 16 + lm) * NPc + hi * 8;

  for (int n0 = 0; n0 < NPc; n0 += 32) {
    v16h a0 = load_a_frag(arow0 + n0);
    v16h a1 = load_a_frag(arow1 + n0);
    v16h b0 = load_b_frag(brow0 + n0);
    v16h b1 = load_b_frag(brow1 + n0);
    acc[0][0] = wmma_f16(a0, b0, acc[0][0]);
    acc[0][1] = wmma_f16(a0, b1, acc[0][1]);
    acc[1][0] = wmma_f16(a1, b0, acc[1][0]);
    acc[1][1] = wmma_f16(a1, b1, acc[1][1]);
  }

  float* outp = kvf + (size_t)wave * HD * HD;  // wave == ((t*B+b)*NH + h)
#pragma unroll
  for (int i2 = 0; i2 < 2; ++i2)
#pragma unroll
    for (int j2 = 0; j2 < 2; ++j2)
#pragma unroll
      for (int r = 0; r < 8; ++r)
        outp[(i2 * 16 + r + hi * 8) * HD + j2 * 16 + lm] = acc[i2][j2][r];
}

// -------------------- a = (q @ kv) * 0.125 ; attn LIF ----------------------
// Per wave: one 16-row n-tile x full 32-col head, all 4 time steps (K=32, one
// WMMA step per tile). Spikes written f16 in [T,B,NPn,C] for the proj GEMM.
__global__ void attn_kernel(const _Float16* __restrict__ qc, // [T,B,C,NPc]
                            const float* __restrict__ kvf,   // [T,B,NH,32,32]
                            _Float16* __restrict__ as_h)     // [T,B,NPn,C]
{
  int wave = (int)((blockIdx.x * blockDim.x + threadIdx.x) >> 5);
  int lane = threadIdx.x & 31;
  int nt = wave % NT, h = (wave / NT) % NH, b = wave / (NT * NH);
  int n0 = nt * 16;
  int hi = lane >> 4, lm = lane & 15;

  v8f acc[T_][2];
#pragma unroll
  for (int t = 0; t < T_; ++t)
#pragma unroll
    for (int e = 0; e < 2; ++e) acc[t][e] = (v8f){};

#pragma unroll
  for (int t = 0; t < T_; ++t) {
    // A: rows m = n (gather, stride NPc; n<=207 reads zero pad cols)
    v16h a;
    {
      int n = n0 + lm;
      const _Float16* p =
          qc + (((size_t)t * B_ + b) * C_ + h * HD) * NPc + n;
      int jb = hi * 16;
#pragma unroll
      for (int i = 0; i < 8; ++i) {
        a[i]     = p[(size_t)(jb + i)      * NPc];
        a[i + 8] = p[(size_t)(jb + 8 + i) * NPc];
      }
    }
    const float* kvp = kvf + (((size_t)t * B_ + b) * NH + h) * HD * HD;
#pragma unroll
    for (int et = 0; et < 2; ++et) {
      v16h bf;
      int e = et * 16 + lm;
      int jb = hi * 16;
#pragma unroll
      for (int i = 0; i < 16; ++i) bf[i] = (_Float16)kvp[(jb + i) * HD + e];
      acc[t][et] = wmma_f16(a, bf, acc[t][et]);
    }
  }

#pragma unroll
  for (int et = 0; et < 2; ++et) {
#pragma unroll
    for (int r = 0; r < 8; ++r) {
      int n = n0 + r + hi * 8;
      int c = h * HD + et * 16 + lm;
      float v = 0.f;
#pragma unroll
      for (int t = 0; t < T_; ++t) {
        float y = acc[t][et][r] * 0.125f;        // attn scale
        v = 0.5f * (v + y);
        float s = (v >= 0.5f) ? 1.0f : 0.0f;     // v_threshold = 0.5
        if (n < N_)
          as_h[(((size_t)t * B_ + b) * NPn + n) * C_ + c] = (_Float16)s;
        v *= (1.0f - s);
      }
    }
  }
}

// ----------------------- proj: 1x1 conv + bias + BN ------------------------
__global__ void proj_kernel(const _Float16* __restrict__ W,
                            const _Float16* __restrict__ as_h, // [T,B,NPn,C]
                            const float* __restrict__ inv_c,
                            const float* __restrict__ beta_c,
                            float* __restrict__ out)           // [T,B,C,N]
{
  int wave = (int)((blockIdx.x * blockDim.x + threadIdx.x) >> 5);
  int lane = threadIdx.x & 31;
  int nt = wave % NT, ot = (wave / NT) % OT;
  int b = (wave / (NT * OT)) % B_, t = wave / (NT * OT * B_);
  int o0 = ot * 16, n0 = nt * 16;
  int hi = lane >> 4, lm = lane & 15;
  int n = n0 + lm;

  v8f acc = (v8f){};
  const _Float16* wrow = W + (size_t)(o0 + lm) * C_ + hi * 8;
  const _Float16* brow =
      as_h + (((size_t)t * B_ + b) * NPn + n) * C_ + hi * 8;

  for (int k0 = 0; k0 < C_; k0 += 32)
    acc = wmma_f16(load_a_frag(wrow + k0), load_b_frag(brow + k0), acc);

  if (n < N_) {
#pragma unroll
    for (int r = 0; r < 8; ++r) {
      int o = o0 + r + hi * 8;
      out[(((size_t)t * B_ + b) * C_ + o) * N_ + n] =
          acc[r] * inv_c[o] + beta_c[o];
    }
  }
}

// ---------------------------------------------------------------------------
extern "C" void kernel_launch(void* const* d_in, const int* in_sizes, int n_in,
                              void* d_out, int out_size, void* d_ws, size_t ws_size,
                              hipStream_t stream) {
  (void)in_sizes; (void)n_in; (void)out_size; (void)ws_size;

  const float* x      = (const float*)d_in[0];
  const float* q_w    = (const float*)d_in[1];
  const float* q_sc   = (const float*)d_in[2];
  const float* q_bi   = (const float*)d_in[3];
  const float* q_me   = (const float*)d_in[4];
  const float* q_va   = (const float*)d_in[5];
  const float* k_w    = (const float*)d_in[6];
  const float* k_sc   = (const float*)d_in[7];
  const float* k_bi   = (const float*)d_in[8];
  const float* k_me   = (const float*)d_in[9];
  const float* k_va   = (const float*)d_in[10];
  const float* v_w    = (const float*)d_in[11];
  const float* v_sc   = (const float*)d_in[12];
  const float* v_bi   = (const float*)d_in[13];
  const float* v_me   = (const float*)d_in[14];
  const float* v_va   = (const float*)d_in[15];
  const float* proj_w = (const float*)d_in[16];
  const float* proj_b = (const float*)d_in[17];
  const float* p_sc   = (const float*)d_in[18];
  const float* p_bi   = (const float*)d_in[19];
  const float* p_me   = (const float*)d_in[20];
  const float* p_va   = (const float*)d_in[21];

  // Output tuple slices: (out, xs, qc, kc, vc), each TBCN f32.
  float* out_y = (float*)d_out;
  float* xs_f  = out_y + TBCN;
  float* qc_f  = out_y + 2 * TBCN;
  float* kc_f  = out_y + 3 * TBCN;
  float* vc_f  = out_y + 4 * TBCN;

  // Workspace layout (all region sizes are multiples of 256B).
  constexpr size_t SZ_NC = (size_t)T_ * B_ * NPn * C_ * 2;  // 20,447,232 B
  constexpr size_t SZ_CN = (size_t)T_ * B_ * C_ * NPc * 2;  // 22,020,096 B
  char* ws = (char*)d_ws;
  size_t off = 0;
  _Float16* xs_h = (_Float16*)(ws + off); off += SZ_NC;     // [T,B,NPn,C]
  _Float16* qc_h = (_Float16*)(ws + off); off += SZ_CN;     // [T,B,C,NPc]
  _Float16* kc_h = (_Float16*)(ws + off); off += SZ_CN;     // [T,B,C,NPc]
  _Float16* vc_h = (_Float16*)(ws + off); off += SZ_CN;     // [T,B,C,NPc]
  _Float16* as_h = (_Float16*)(ws + off); off += SZ_NC;     // [T,B,NPn,C]
  float*    kvf  = (float*)(ws + off);    off += (size_t)T_ * B_ * NH * HD * HD * 4;
  _Float16* whq  = (_Float16*)(ws + off); off += (size_t)C_ * C_ * 2;
  _Float16* whk  = (_Float16*)(ws + off); off += (size_t)C_ * C_ * 2;
  _Float16* whv  = (_Float16*)(ws + off); off += (size_t)C_ * C_ * 2;
  _Float16* whp  = (_Float16*)(ws + off); off += (size_t)C_ * C_ * 2;
  float*    cst  = (float*)(ws + off);    off += 8 * C_ * 4;

  const int BLK = 256;  // 8 wave32 per workgroup

  // 0) zero the pad regions so GEMM fragment loads need no guards
  {
    int tot_nc = T_ * B_ * (NPn - N_) * C_;     // 589,824 -> 2304 blocks
    int tot_cn = T_ * B_ * C_ * (NPc - N_);     // 1,376,256 -> 5376 blocks
    zero_pad_nc_kernel<<<tot_nc / BLK, BLK, 0, stream>>>(xs_h);
    zero_pad_nc_kernel<<<tot_nc / BLK, BLK, 0, stream>>>(as_h);
    zero_pad_cn_kernel<<<tot_cn / BLK, BLK, 0, stream>>>(qc_h);
    zero_pad_cn_kernel<<<tot_cn / BLK, BLK, 0, stream>>>(kc_h);
    zero_pad_cn_kernel<<<tot_cn / BLK, BLK, 0, stream>>>(vc_h);
  }

  // 1) weight f16 conversion + BN constant folding
  {
    int total = 4 * C_ * C_ + 4 * C_;           // 591360 -> 2310 blocks exactly
    prep_kernel<<<(total + BLK - 1) / BLK, BLK, 0, stream>>>(
        q_w, k_w, v_w, proj_w,
        q_sc, q_bi, q_me, q_va,
        k_sc, k_bi, k_me, k_va,
        v_sc, v_bi, v_me, v_va,
        proj_b, p_sc, p_bi, p_me, p_va,
        whq, whk, whv, whp, cst);
  }

  // 2) proj_lif over input
  lif_x_kernel<<<(int)((BCN + BLK - 1) / BLK), BLK, 0, stream>>>(x, xs_f, xs_h);

  // 3) q/k/v conv+BN+LIF (fused, WMMA)
  {
    int waves = B_ * OT * NT;                   // 9984 -> 1248 blocks exactly
    dim3 g(waves / 8);
    qkv_gemm_lif_kernel<<<g, BLK, 0, stream>>>(whq, xs_h, cst + 0 * C_, cst + 1 * C_, qc_f, qc_h);
    qkv_gemm_lif_kernel<<<g, BLK, 0, stream>>>(whk, xs_h, cst + 2 * C_, cst + 3 * C_, kc_f, kc_h);
    qkv_gemm_lif_kernel<<<g, BLK, 0, stream>>>(whv, xs_h, cst + 4 * C_, cst + 5 * C_, vc_f, vc_h);
  }

  // 4) kv = k^T v per (t,b,h)
  kv_kernel<<<(T_ * B_ * NH) / 8, BLK, 0, stream>>>(kc_h, vc_h, kvf);

  // 5) a = q@kv * 0.125, attn LIF (vth=0.5)
  attn_kernel<<<(B_ * NH * NT) / 8, BLK, 0, stream>>>(qc_h, kvf, as_h);

  // 6) proj conv + bias + BN -> output slice 0
  proj_kernel<<<(T_ * B_ * OT * NT) / 8, BLK, 0, stream>>>(
      whp, as_h, cst + 6 * C_, cst + 7 * C_, out_y);
}
